// RetSA_3685081940152
// MI455X (gfx1250) — matrix operand order repaired and checked by reading, not compiled, not run on hardware
//
#include <hip/hip_runtime.h>

// ---------------------------------------------------------------------------
// RetSA forward for MI455X (gfx1250, wave32, WMMA bf16 16x16x32, f32 accum)
// rel_pos (170 MB, single-use) is streamed with non-temporal loads + prefetch
// so the 144x-reused K/V heads stay resident in the 192 MB L2.
// ---------------------------------------------------------------------------

typedef __attribute__((ext_vector_type(16))) __bf16 v16bf;
typedef __attribute__((ext_vector_type(8)))  float  v8f;
typedef __attribute__((ext_vector_type(4)))  float  v4f;

#define Bb   2
#define Hh   48
#define Wdim 48
#define C    256
#define NH   8
#define DK   32
#define DV   32
#define L    2304          /* Hh*Wdim */
#define MTOT 4608          /* Bb*L    */

__constant__ float kSCALING = 0.17677669529663687f;  /* 32^-0.5 */
#define LOG2E 1.4426950408889634f

__device__ __forceinline__ v8f zero8() {
  v8f z;
#pragma unroll
  for (int i = 0; i < 8; ++i) z[i] = 0.0f;
  return z;
}

__device__ __forceinline__ __bf16 tobf(float f) { return (__bf16)f; }

// Load a 16-element bf16 fragment from two 16-byte-aligned 8-element pieces.
__device__ __forceinline__ v16bf ld_frag(const __bf16* p0, const __bf16* p1) {
  union { uint4 q[2]; v16bf v; } u;
  u.q[0] = *reinterpret_cast<const uint4*>(p0);
  u.q[1] = *reinterpret_cast<const uint4*>(p1);
  return u.v;
}

__device__ __forceinline__ v8f wmma_bf16(v16bf a, v16bf b, v8f c) {
  // v_wmma_f32_16x16x32_bf16  D = A*B + C
  return __builtin_amdgcn_wmma_f32_16x16x32_bf16(
      false, a, false, b, (short)0, c, false, false);
}

// cross-lane helpers (wave32)
__device__ __forceinline__ float xor16(float v) {
  // ds_swizzle group-of-32: and=0x1f, or=0, xor=0x10  -> 0x401F (SWAPX16)
  return __int_as_float(__builtin_amdgcn_ds_swizzle(__float_as_int(v), 0x401F));
}
__device__ __forceinline__ float lane_bcast(float v, int src) {
  return __int_as_float(__builtin_amdgcn_ds_bpermute(src << 2, __float_as_int(v)));
}

// ---------------------------------------------------------------------------
// Pass 0a: x (f32) -> bf16
__global__ void cvt_x(const float* __restrict__ x, __bf16* __restrict__ xb, int n) {
  int i = blockIdx.x * 256 + threadIdx.x;
  if (i < n) xb[i] = tobf(x[i]);
}

// Pass 0b: W{q,k,v,o} (f32, [k][n]) -> bf16 transposed [n][k] (B-fragment wants
// per-lane contiguous K at fixed N).
__global__ void transpose_w(const float* __restrict__ Wq, const float* __restrict__ Wk,
                            const float* __restrict__ Wv, const float* __restrict__ Wo,
                            __bf16* __restrict__ Wt) {
  int k = threadIdx.x;          // 0..255
  int n = blockIdx.x;           // 0..255
  int z = blockIdx.y;           // 0..3
  const float* W = (z == 0) ? Wq : (z == 1) ? Wk : (z == 2) ? Wv : Wo;
  Wt[(size_t)z * C * C + (size_t)n * C + k] = tobf(W[(size_t)k * C + n]);
}

// ---------------------------------------------------------------------------
// WMMA GEMM: out(M=4608 x N=256) = A(bf16, row-major [m][k]) * Wt^T + bias.
// Block = 256 threads = 8 waves; block covers 32 rows x 256 cols; each wave a
// 32x32 tile (2x2 WMMA accumulators). modeBase+blockIdx.y selects weights:
//   0: Q -> q_bf (bf16)            1: K -> k_bf (bf16, *SCALING)
//   2: V -> v_f32 + v_t (bf16, [b][h][d][L])     3: final -> outF (f32)
__global__ void gemm_bf16(const __bf16* __restrict__ A, const __bf16* __restrict__ WtAll,
                          const float* __restrict__ bq, const float* __restrict__ bk,
                          const float* __restrict__ bv, const float* __restrict__ bo,
                          int modeBase,
                          __bf16* __restrict__ q_bf, __bf16* __restrict__ k_bf,
                          __bf16* __restrict__ v_t,  float* __restrict__ v_f32,
                          float* __restrict__ outF) {
  const int mode = modeBase + blockIdx.y;
  const __bf16* Wt = WtAll + (size_t)mode * C * C;
  const float* bias = (mode == 0) ? bq : (mode == 1) ? bk : (mode == 2) ? bv : bo;

  const int lane = threadIdx.x & 31;
  const int wv   = threadIdx.x >> 5;       // 0..7 -> n-tile
  const int m0   = blockIdx.x * 32;
  const int n0   = wv * 32;
  const int lm   = lane & 15;
  const int g    = lane >> 4;

  v8f acc[2][2];
  acc[0][0] = zero8(); acc[0][1] = zero8();
  acc[1][0] = zero8(); acc[1][1] = zero8();

  for (int kk = 0; kk < C; kk += 32) {
    v16bf afr[2], bfr[2];
#pragma unroll
    for (int ms = 0; ms < 2; ++ms) {
      const __bf16* pa = A + (size_t)(m0 + ms * 16 + lm) * C + kk;
      afr[ms] = ld_frag(pa + g * 8, pa + 16 + g * 8);          // A 16x32 layout
    }
#pragma unroll
    for (int ns = 0; ns < 2; ++ns) {
      const __bf16* pb = Wt + (size_t)(n0 + ns * 16 + lm) * C + kk;
      bfr[ns] = ld_frag(pb + g * 8, pb + 16 + g * 8);          // B 32x16 layout
    }
#pragma unroll
    for (int ms = 0; ms < 2; ++ms)
#pragma unroll
      for (int ns = 0; ns < 2; ++ns)
        acc[ms][ns] = wmma_bf16(afr[ms], bfr[ns], acc[ms][ns]);
  }

#pragma unroll
  for (int ns = 0; ns < 2; ++ns) {
    const int n  = n0 + ns * 16 + lm;
    const float bb = bias[n];
#pragma unroll
    for (int ms = 0; ms < 2; ++ms) {
#pragma unroll
      for (int r = 0; r < 8; ++r) {
        const int m = m0 + ms * 16 + r + 8 * g;   // C/D layout: M = r + 8*g
        float val = acc[ms][ns][r] + bb;
        if (mode == 0) {
          q_bf[(size_t)m * C + n] = tobf(val);
        } else if (mode == 1) {
          k_bf[(size_t)m * C + n] = tobf(val * kSCALING);
        } else if (mode == 2) {
          v_f32[(size_t)m * C + n] = val;
          const int bI = (m >= L) ? 1 : 0;
          const int mr = m - bI * L;
          const int hh = n >> 5, dd = n & 31;
          v_t[(((size_t)bI * NH + hh) * DV + dd) * L + mr] = tobf(val);
        } else {
          outF[(size_t)m * C + n] = val;
        }
      }
    }
  }
}

// ---------------------------------------------------------------------------
// Flash attention, one wave per (b, head, 16-row m-tile). Computes S^T = K*Q^T
// so softmax rows live on lane&15 and exp(S^T) is *already* in the A-fragment
// layout needed for P@V. rel_pos is streamed non-temporally (single use).
__global__ void attn_kernel(const __bf16* __restrict__ q_bf, const __bf16* __restrict__ k_bf,
                            const __bf16* __restrict__ v_t, const float* __restrict__ rel,
                            float* __restrict__ attnO) {
  const int lane = threadIdx.x & 31;
  const int wv   = threadIdx.x >> 5;
  const int idx  = blockIdx.x * 8 + wv;     // 0 .. 2*8*144-1
  const int bI   = idx / (NH * 144);
  const int h    = (idx / 144) & (NH - 1);
  const int mt   = idx % 144;
  const int m0   = mt * 16;
  const int lm   = lane & 15;
  const int g    = lane >> 4;
  const int base = g * 8;                   // first source lane for D-row bcast

  // Q^T B-fragment (loop invariant): lane holds column m=m0+lm, K=d in A-table order
  const __bf16* pq = q_bf + (size_t)(bI * L + m0 + lm) * C + h * DK;
  const v16bf qf = ld_frag(pq + g * 8, pq + 16 + g * 8);

  const float*  relrow = rel + ((size_t)h * L + (m0 + lm)) * L;
  const __bf16* vbase  = v_t + ((size_t)bI * NH + h) * DV * L;

  v8f accO[2]; accO[0] = zero8(); accO[1] = zero8();
  float Mrun = -INFINITY, Srun = 0.0f;

  for (int n0 = 0; n0 < L; n0 += 32) {
    // prefetch next iteration's rel_pos chunk for this lane (streamed NT)
    if (n0 + 32 < L) __builtin_prefetch(relrow + n0 + 32 + g * 8, 0, 0);

    float sS[2][8];
    float tmax = -INFINITY;
#pragma unroll
    for (int c = 0; c < 2; ++c) {
      // K A-fragment: rows n = n0+16c+lm, K=d (full DK=32 in one WMMA)
      const __bf16* pk = k_bf + (size_t)(bI * L + n0 + c * 16 + lm) * C + h * DK;
      const v16bf kf = ld_frag(pk + g * 8, pk + 16 + g * 8);
      v8f s = wmma_bf16(kf, qf, zero8());        // S^T chunk: M=n, N=m
      // rel_pos: element (m=m0+lm, n=n0+16c+8g+r) -> 8 consecutive floats/lane
      const v4f* pr = reinterpret_cast<const v4f*>(relrow + n0 + c * 16 + g * 8);
      const v4f r0 = __builtin_nontemporal_load(pr);       // TH=NT: single use
      const v4f r1 = __builtin_nontemporal_load(pr + 1);
      s[0] += r0[0]; s[1] += r0[1]; s[2] += r0[2]; s[3] += r0[3];
      s[4] += r1[0]; s[5] += r1[1]; s[6] += r1[2]; s[7] += r1[3];
#pragma unroll
      for (int r = 0; r < 8; ++r) { sS[c][r] = s[r]; tmax = fmaxf(tmax, s[r]); }
    }
    // full row max across the 32 columns: lanes l and l^16 hold the two halves
    tmax = fmaxf(tmax, xor16(tmax));
    const float newM = fmaxf(Mrun, tmax);
    const float corr = __builtin_amdgcn_exp2f((Mrun - newM) * LOG2E);

    float lsum = 0.0f;
    v16bf pf;                                    // P as A-fragment, for free:
#pragma unroll
    for (int c = 0; c < 2; ++c)
#pragma unroll
      for (int r = 0; r < 8; ++r) {
        const float p = __builtin_amdgcn_exp2f((sS[c][r] - newM) * LOG2E);
        lsum += p;
        pf[c * 8 + r] = tobf(p);                 // K = 16c + 8g + r  (A layout)
      }
    lsum += xor16(lsum);
    Srun = Srun * corr + lsum;
    Mrun = newM;

    // rescale output accumulators: D rows are M = r + 8g -> broadcast corr
#pragma unroll
    for (int r = 0; r < 8; ++r) {
      const float cr = lane_bcast(corr, base + r);
      accO[0][r] *= cr;
      accO[1][r] *= cr;
    }
    // P(16x32) @ V(32x32): two WMMAs, one per 16-wide d-chunk
#pragma unroll
    for (int dch = 0; dch < 2; ++dch) {
      const __bf16* pv = vbase + (size_t)(dch * 16 + lm) * L + n0;
      const v16bf vf = ld_frag(pv + g * 8, pv + 16 + g * 8);  // B: lane=col d, K=n in A-table order
      accO[dch] = wmma_bf16(pf, vf, accO[dch]);
    }
  }

  const float rinv = 1.0f / Srun;
#pragma unroll
  for (int r = 0; r < 8; ++r) {
    const float rr = lane_bcast(rinv, base + r);
    const int m = m0 + r + 8 * g;
    float* po = attnO + (size_t)(bI * L + m) * C + h * DV;
    po[lm]      = accO[0][r] * rr;
    po[16 + lm] = accO[1][r] * rr;
  }
}

// ---------------------------------------------------------------------------
// Depthwise 5x5 conv (stride 1, pad 2) over v_f32 (b,48,48,256) + bias.
__global__ void lepe_conv(const float* __restrict__ v, const float* __restrict__ Wl,
                          const float* __restrict__ bl, float* __restrict__ lepe) {
  const int c  = threadIdx.x;                  // channel
  const int sp = blockIdx.x;                   // b*2304 + y*48 + x
  const int b  = sp / L;
  const int yx = sp - b * L;
  const int y  = yx / Wdim;
  const int x  = yx - y * Wdim;
  float acc = bl[c];
#pragma unroll
  for (int ky = 0; ky < 5; ++ky) {
    const int yy = y + ky - 2;
    if (yy < 0 || yy >= Hh) continue;
#pragma unroll
    for (int kx = 0; kx < 5; ++kx) {
      const int xx = x + kx - 2;
      if (xx < 0 || xx >= Wdim) continue;
      acc += v[((size_t)b * L + yy * Wdim + xx) * C + c] * Wl[(ky * 5 + kx) * C + c];
    }
  }
  lepe[(size_t)sp * C + c] = acc;
}

// Pass: s_bf = bf16(attn + lepe)
__global__ void fuse_add(const float* __restrict__ a, const float* __restrict__ b,
                         __bf16* __restrict__ s, int n) {
  int i = blockIdx.x * 256 + threadIdx.x;
  if (i < n) s[i] = tobf(a[i] + b[i]);
}

// ---------------------------------------------------------------------------
extern "C" void kernel_launch(void* const* d_in, const int* in_sizes, int n_in,
                              void* d_out, int out_size, void* d_ws, size_t ws_size,
                              hipStream_t stream) {
  const float* x   = (const float*)d_in[0];
  const float* rel = (const float*)d_in[1];
  const float* Wq  = (const float*)d_in[2];
  const float* bq  = (const float*)d_in[3];
  const float* Wk  = (const float*)d_in[4];
  const float* bk  = (const float*)d_in[5];
  const float* Wv  = (const float*)d_in[6];
  const float* bv  = (const float*)d_in[7];
  const float* Wl  = (const float*)d_in[8];
  const float* bl  = (const float*)d_in[9];
  const float* Wo  = (const float*)d_in[10];
  const float* bo  = (const float*)d_in[11];
  float* out = (float*)d_out;

  char* p = (char*)d_ws;
  auto take = [&](size_t bytes) -> char* {
    char* r = p;
    p += (bytes + 255) & ~(size_t)255;
    return r;
  };
  __bf16* x_bf  = (__bf16*)take((size_t)MTOT * C * 2);
  __bf16* Wt    = (__bf16*)take((size_t)4 * C * C * 2);
  __bf16* q_bf  = (__bf16*)take((size_t)MTOT * C * 2);
  __bf16* k_bf  = (__bf16*)take((size_t)MTOT * C * 2);
  __bf16* v_t   = (__bf16*)take((size_t)MTOT * C * 2);
  float*  v_f32 = (float*)take((size_t)MTOT * C * 4);
  float*  lepe  = (float*)take((size_t)MTOT * C * 4);
  float*  attnO = (float*)take((size_t)MTOT * C * 4);
  __bf16* s_bf  = (__bf16*)take((size_t)MTOT * C * 2);

  const int NE = MTOT * C;  // 1179648

  cvt_x<<<NE / 256, 256, 0, stream>>>(x, x_bf, NE);
  transpose_w<<<dim3(256, 4), 256, 0, stream>>>(Wq, Wk, Wv, Wo, Wt);
  gemm_bf16<<<dim3(144, 3), 256, 0, stream>>>(x_bf, Wt, bq, bk, bv, bo, 0,
                                              q_bf, k_bf, v_t, v_f32, nullptr);
  attn_kernel<<<288, 256, 0, stream>>>(q_bf, k_bf, v_t, rel, attnO);
  lepe_conv<<<MTOT, 256, 0, stream>>>(v_f32, Wl, bl, lepe);
  fuse_add<<<NE / 256, 256, 0, stream>>>(attnO, lepe, s_bf, NE);
  gemm_bf16<<<dim3(144, 1), 256, 0, stream>>>(s_bf, Wt, bq, bk, bv, bo, 3,
                                              nullptr, nullptr, nullptr, nullptr, out);
}
